// QuantumAdaGN_12524124635733
// MI455X (gfx1250) — matrix operand order repaired and checked by reading, not compiled, not run on hardware
//
#include <hip/hip_runtime.h>
#include <hip/hip_bf16.h>
#include <math.h>

// ---------------------------------------------------------------------------
// QuantumAdaGN fused implementation for gfx1250 (MI455X).
//   y = normalize(Win x + b);  z = U_b y (complex);  p = |z|^2;  out = Wout p + b
// U_b (64x64 complex) is per-batch only (style broadcast over H,W), so it is
// precomputed once per batch and the hot path is 3 fused bf16 WMMA GEMMs.
// ---------------------------------------------------------------------------

typedef float  v8f   __attribute__((ext_vector_type(8)));
typedef __bf16 v16bf __attribute__((ext_vector_type(16)));

#define NQ   6
#define QDIM 64     // 2^NQ
#define CCH  256
#define SD   512
#define HW   4096   // 64*64
#define NB   16

// workspace layout (bytes)
#define WS_WINFRAG  0                       // 64x256  bf16 frags : 32768 B
#define WS_WOUTFRAG 32768                   // 256x64  bf16 frags : 32768 B
#define WS_UREFRAG  65536                   // 16 x 64x64 bf16    : 131072 B
#define WS_UIMFRAG  (65536 + 131072)        // 16 x 64x64 bf16    : 131072 B

__device__ __forceinline__ v8f vzero8() {
    v8f v;
#pragma unroll
    for (int i = 0; i < 8; i++) v[i] = 0.0f;
    return v;
}

// A-fragment element -> (m,k) mapping used by this code's bf16 A frags (contiguous k, B reads match).
// lane l: m = l&15 ; elem e -> k = (l>>4)*16 + e
__device__ __forceinline__ int afrag_k(int lane, int e) {
    return (lane >> 4) * 16 + e;
}

// ---------------------------------------------------------------------------
// Kernel 1: convert Win (64x256) and Wout (256x64) f32 -> bf16 A-fragments.
// frag storage: ((mt*KC + kc)*32 + lane)*16 + e
// ---------------------------------------------------------------------------
__global__ void setup_weights_kernel(const float* __restrict__ win,
                                     const float* __restrict__ wout,
                                     __bf16* __restrict__ winfrag,
                                     __bf16* __restrict__ woutfrag) {
    int t = threadIdx.x;
    // Win: MT=4, KC=8  -> 16384 elements
    for (int idx = t; idx < 16384; idx += 256) {
        int e = idx & 15, lane = (idx >> 4) & 31, frag = idx >> 9;
        int kc = frag & 7, mt = frag >> 3;
        int m = mt * 16 + (lane & 15);
        int k = kc * 32 + afrag_k(lane, e);
        winfrag[idx] = (__bf16)win[m * CCH + k];
    }
    // Wout: MT=16, KC=2 -> 16384 elements
    for (int idx = t; idx < 16384; idx += 256) {
        int e = idx & 15, lane = (idx >> 4) & 31, frag = idx >> 9;
        int kc = frag & 1, mt = frag >> 1;
        int m = mt * 16 + (lane & 15);
        int k = kc * 32 + afrag_k(lane, e);
        woutfrag[idx] = (__bf16)wout[m * QDIM + k];
    }
}

// ---------------------------------------------------------------------------
// Kernel 2: per-batch circuit unitary. One block per batch, 64 threads
// (thread = column of U). Columns evolve independently through the gates.
// qubit i corresponds to bit (5-i) of the flat 64-dim index.
// ---------------------------------------------------------------------------
__global__ void setup_unitary_kernel(const float* __restrict__ style,
                                     const float* __restrict__ smw,
                                     const float* __restrict__ smb,
                                     const float* __restrict__ qw,
                                     __bf16* __restrict__ urefrag,
                                     __bf16* __restrict__ uimfrag) {
    __shared__ float cth[18], sth[18];
    __shared__ float ure[QDIM][QDIM];
    __shared__ float uim[QDIM][QDIM];
    const int b = blockIdx.x;
    const int t = threadIdx.x;

    if (t < 18) {
        float acc = 0.0f;
        const float* sv = style + b * SD;
        const float* wr = smw + t * SD;
        for (int j = 0; j < SD; j++) acc += sv[j] * wr[j];
        float ang = 0.5f * (qw[t] + acc + smb[t]);
        cth[t] = cosf(ang);
        sth[t] = sinf(ang);
    }
    // init identity column c = t
    const int c = t;
    for (int r = 0; r < QDIM; r++) {
        ure[r][c] = (r == c) ? 1.0f : 0.0f;
        uim[r][c] = 0.0f;
    }
    __syncthreads();

    // RX then RY per qubit
    for (int i = 0; i < NQ; i++) {
        int mask = 1 << (5 - i);
        // RX(theta_i0): [[c,-is],[-is,c]]
        {
            float ch = cth[i * 3 + 0], sh = sth[i * 3 + 0];
            for (int pr = 0; pr < 32; pr++) {
                int low = pr & (mask - 1);
                int high = (pr & ~(mask - 1)) << 1;
                int r0 = high | low, r1 = r0 | mask;
                float ar = ure[r0][c], ai = uim[r0][c];
                float br = ure[r1][c], bi = uim[r1][c];
                ure[r0][c] = ch * ar + sh * bi;  uim[r0][c] = ch * ai - sh * br;
                ure[r1][c] = sh * ai + ch * br;  uim[r1][c] = -sh * ar + ch * bi;
            }
        }
        // RY(theta_i1): [[c,-s],[s,c]]
        {
            float ch = cth[i * 3 + 1], sh = sth[i * 3 + 1];
            for (int pr = 0; pr < 32; pr++) {
                int low = pr & (mask - 1);
                int high = (pr & ~(mask - 1)) << 1;
                int r0 = high | low, r1 = r0 | mask;
                float ar = ure[r0][c], ai = uim[r0][c];
                float br = ure[r1][c], bi = uim[r1][c];
                ure[r0][c] = ch * ar - sh * br;  uim[r0][c] = ch * ai - sh * bi;
                ure[r1][c] = sh * ar + ch * br;  uim[r1][c] = sh * ai + ch * bi;
            }
        }
    }
    // CNOT chain: control i, target i+1
    for (int i = 0; i < NQ - 1; i++) {
        int cm = 1 << (5 - i), tm = 1 << (4 - i);
        for (int r = 0; r < QDIM; r++) {
            if ((r & cm) && !(r & tm)) {
                int r1 = r | tm;
                float a = ure[r][c]; ure[r][c] = ure[r1][c]; ure[r1][c] = a;
                float bqi = uim[r][c]; uim[r][c] = uim[r1][c]; uim[r1][c] = bqi;
            }
        }
    }
    // final RY layer
    for (int i = 0; i < NQ; i++) {
        int mask = 1 << (5 - i);
        float ch = cth[i * 3 + 2], sh = sth[i * 3 + 2];
        for (int pr = 0; pr < 32; pr++) {
            int low = pr & (mask - 1);
            int high = (pr & ~(mask - 1)) << 1;
            int r0 = high | low, r1 = r0 | mask;
            float ar = ure[r0][c], ai = uim[r0][c];
            float br = ure[r1][c], bi = uim[r1][c];
            ure[r0][c] = ch * ar - sh * br;  uim[r0][c] = ch * ai - sh * bi;
            ure[r1][c] = sh * ar + ch * br;  uim[r1][c] = sh * ai + ch * bi;
        }
    }
    __syncthreads();

    // emit bf16 A-fragments (MT=4, KC=2)
    __bf16* ureb = urefrag + b * (QDIM * QDIM);
    __bf16* uimb = uimfrag + b * (QDIM * QDIM);
    for (int idx = t; idx < QDIM * QDIM; idx += 64) {
        int e = idx & 15, lane = (idx >> 4) & 31, frag = idx >> 9;
        int kc = frag & 1, mt = frag >> 1;
        int m = mt * 16 + (lane & 15);
        int k = kc * 32 + afrag_k(lane, e);
        ureb[idx] = (__bf16)ure[m][k];
        uimb[idx] = (__bf16)uim[m][k];
    }
}

// ---------------------------------------------------------------------------
// Kernel 3: fused main path. 1024 blocks (16 batches x 64 pixel tiles),
// 128 threads = 4 waves; each wave owns a 16-pixel WMMA column tile.
// ---------------------------------------------------------------------------
#define XP 272   // x tile LDS pitch (elements, 32B-aligned rows)
#define YP 80    // y/p tile LDS pitch

__global__ __launch_bounds__(128)
void qagn_main_kernel(const float* __restrict__ x,
                      const float* __restrict__ inb_g,
                      const float* __restrict__ outb_g,
                      const __bf16* __restrict__ winfrag,
                      const __bf16* __restrict__ woutfrag,
                      const __bf16* __restrict__ urefrag,
                      const __bf16* __restrict__ uimfrag,
                      float* __restrict__ out) {
    __shared__ __align__(32) __bf16 xs[64 * XP];
    __shared__ __align__(32) __bf16 yb[64 * YP];
    __shared__ __align__(32) __bf16 pbuf[64 * YP];
    __shared__ float inb[QDIM];
    __shared__ float outb[CCH];

    const int blk = blockIdx.x;
    const int b = blk >> 6;
    const int pixbase = (blk & 63) * 64;
    const int tid = threadIdx.x;

    if (tid < QDIM) inb[tid] = inb_g[tid];
    outb[tid] = outb_g[tid];
    outb[tid + 128] = outb_g[tid + 128];

    // stage x tile [64 px x 256 ch] -> LDS bf16, [p][c] layout.
    // float4 loads: fully coalesced global_load_b128 across the wave.
    const float4* xb4 = (const float4*)(x + (size_t)b * CCH * HW + pixbase);
    for (int i = tid; i < CCH * 16; i += 128) {
        int q = i & 15;            // float4 index within the 64-pixel row
        int cch = i >> 4;          // channel
        float4 v = xb4[(size_t)cch * (HW / 4) + q];
        int p4 = q * 4;
        xs[(p4 + 0) * XP + cch] = (__bf16)v.x;
        xs[(p4 + 1) * XP + cch] = (__bf16)v.y;
        xs[(p4 + 2) * XP + cch] = (__bf16)v.z;
        xs[(p4 + 3) * XP + cch] = (__bf16)v.w;
    }
    __syncthreads();

    const int lane = tid & 31;
    const int wave = tid >> 5;
    const int nl = lane & 15;
    const int hi = lane >> 4;     // 0 or 1
    const int p = wave * 16 + nl; // pixel within tile owned by this lane

    // ---------------- G1: y = Win x + b (M=64, K=256) ----------------
    v8f acc[4];
#pragma unroll
    for (int mt = 0; mt < 4; mt++) acc[mt] = vzero8();
#pragma unroll
    for (int kc = 0; kc < 8; kc++) {
        v16bf bfrag = *(const v16bf*)&xs[p * XP + kc * 32 + hi * 16];
#pragma unroll
        for (int mt = 0; mt < 4; mt++) {
            v16bf afrag = *(const v16bf*)&winfrag[((mt * 8 + kc) * 32 + lane) * 16];
            acc[mt] = __builtin_amdgcn_wmma_f32_16x16x32_bf16(
                false, afrag, false, bfrag, (short)0, acc[mt], false, false);
        }
    }
    // bias + per-pixel L2 norm (features split across lane and lane^16)
    float sumsq = 0.0f;
#pragma unroll
    for (int mt = 0; mt < 4; mt++)
#pragma unroll
        for (int i = 0; i < 8; i++) {
            float v = acc[mt][i] + inb[mt * 16 + hi * 8 + i];
            acc[mt][i] = v;
            sumsq += v * v;
        }
    sumsq += __shfl_xor(sumsq, 16, 32);
    float scale = 1.0f / (sqrtf(sumsq) + 1e-9f);
#pragma unroll
    for (int mt = 0; mt < 4; mt++)
#pragma unroll
        for (int i = 0; i < 8; i++)
            yb[p * YP + mt * 16 + hi * 8 + i] = (__bf16)(acc[mt][i] * scale);
    __syncthreads();

    // ---------------- G2: z = U_b y (complex, M=64, K=64) ----------------
    v8f zre[4], zim[4];
#pragma unroll
    for (int mt = 0; mt < 4; mt++) { zre[mt] = vzero8(); zim[mt] = vzero8(); }
    const __bf16* ureb = urefrag + b * (QDIM * QDIM);
    const __bf16* uimb = uimfrag + b * (QDIM * QDIM);
#pragma unroll
    for (int kc = 0; kc < 2; kc++) {
        v16bf bfrag = *(const v16bf*)&yb[p * YP + kc * 32 + hi * 16];
#pragma unroll
        for (int mt = 0; mt < 4; mt++) {
            v16bf ar = *(const v16bf*)&ureb[((mt * 2 + kc) * 32 + lane) * 16];
            v16bf ai = *(const v16bf*)&uimb[((mt * 2 + kc) * 32 + lane) * 16];
            zre[mt] = __builtin_amdgcn_wmma_f32_16x16x32_bf16(
                false, ar, false, bfrag, (short)0, zre[mt], false, false);
            zim[mt] = __builtin_amdgcn_wmma_f32_16x16x32_bf16(
                false, ai, false, bfrag, (short)0, zim[mt], false, false);
        }
    }
    // probabilities p = |z|^2
#pragma unroll
    for (int mt = 0; mt < 4; mt++)
#pragma unroll
        for (int i = 0; i < 8; i++) {
            float pr = zre[mt][i] * zre[mt][i] + zim[mt][i] * zim[mt][i];
            pbuf[p * YP + mt * 16 + hi * 8 + i] = (__bf16)pr;
        }
    __syncthreads();

    // ---------------- G3: out = Wout p + b (M=256, K=64) ----------------
    v16bf pf0 = *(const v16bf*)&pbuf[p * YP + 0 + hi * 16];
    v16bf pf1 = *(const v16bf*)&pbuf[p * YP + 32 + hi * 16];
    float* ob = out + (size_t)b * CCH * HW + pixbase;
#pragma unroll
    for (int mt = 0; mt < 16; mt++) {
        v8f oacc = vzero8();
        v16bf w0 = *(const v16bf*)&woutfrag[((mt * 2 + 0) * 32 + lane) * 16];
        v16bf w1 = *(const v16bf*)&woutfrag[((mt * 2 + 1) * 32 + lane) * 16];
        oacc = __builtin_amdgcn_wmma_f32_16x16x32_bf16(
            false, w0, false, pf0, (short)0, oacc, false, false);
        oacc = __builtin_amdgcn_wmma_f32_16x16x32_bf16(
            false, w1, false, pf1, (short)0, oacc, false, false);
#pragma unroll
        for (int i = 0; i < 8; i++) {
            int cchan = mt * 16 + hi * 8 + i;
            ob[(size_t)cchan * HW + p] = oacc[i] + outb[cchan];
        }
    }
}

// ---------------------------------------------------------------------------
// launch
// ---------------------------------------------------------------------------
extern "C" void kernel_launch(void* const* d_in, const int* in_sizes, int n_in,
                              void* d_out, int out_size, void* d_ws, size_t ws_size,
                              hipStream_t stream) {
    const float* x     = (const float*)d_in[0];
    const float* style = (const float*)d_in[1];
    const float* winw  = (const float*)d_in[2];
    const float* winb  = (const float*)d_in[3];
    const float* smw   = (const float*)d_in[4];
    const float* smb   = (const float*)d_in[5];
    const float* qw    = (const float*)d_in[6];
    const float* woutw = (const float*)d_in[7];
    const float* woutb = (const float*)d_in[8];
    float* out = (float*)d_out;

    char* ws = (char*)d_ws;
    __bf16* winfrag  = (__bf16*)(ws + WS_WINFRAG);
    __bf16* woutfrag = (__bf16*)(ws + WS_WOUTFRAG);
    __bf16* urefrag  = (__bf16*)(ws + WS_UREFRAG);
    __bf16* uimfrag  = (__bf16*)(ws + WS_UIMFRAG);

    hipLaunchKernelGGL(setup_weights_kernel, dim3(1), dim3(256), 0, stream,
                       winw, woutw, winfrag, woutfrag);
    hipLaunchKernelGGL(setup_unitary_kernel, dim3(NB), dim3(64), 0, stream,
                       style, smw, smb, qw, urefrag, uimfrag);
    hipLaunchKernelGGL(qagn_main_kernel, dim3(NB * 64), dim3(128), 0, stream,
                       x, winb, woutb, winfrag, woutfrag, urefrag, uimfrag, out);
}
